// ScatterBrain_71674414235761
// MI455X (gfx1250) — matrix operand order repaired and checked by reading, not compiled, not run on hardware
//
#include <hip/hip_runtime.h>
#include <hip/hip_bf16.h>

// ---------------------------------------------------------------------------
// Taylor linear attention (Based / zoology) on MI455X gfx1250.
// B=4 L=2048 DM=768 H=12 FD=16 HD=64, phi-dim 273 padded to D=288, CHUNK=128.
// All matmuls via v_wmma_f32_16x16x32_f16 (f16 inputs, f32 accumulate).
// phi layout is PERMUTED to [x(x)x | x/2 | 1 | zeros] (dot-product invariant)
// so every segment is 16-byte aligned for b128 vector access.
// ---------------------------------------------------------------------------

typedef __attribute__((ext_vector_type(16))) _Float16 v16h;
typedef __attribute__((ext_vector_type(8)))  float    v8f;

#define LA_B 4
#define LA_L 2048
#define LA_DM 768
#define LA_H 12
#define LA_FD 16
#define LA_HD 64
#define LA_D 288      // 256 + 16 + 1, padded to 288 (zeros)
#define LA_C 128      // chunk
#define LA_NCH 16     // L / CHUNK

struct HFrag { union { v16h v; _Float16 e[16]; uint4 q[2]; }; };
union H8 { uint4 q; _Float16 e[8]; };

// A-matrix fragment (16x32 f16, straight K layout):
// lane = (row 0..15) + 16*half ; elems 0..15 -> K = kb+16*half+i
__device__ __forceinline__ v16h ldfragA_h(const _Float16* p, int ld, int rb, int kb, int lane) {
  const int r = rb + (lane & 15), hf = lane >> 4;
  const _Float16* base = p + (size_t)r * ld + kb + 16 * hf;
  HFrag f;
  f.q[0] = *reinterpret_cast<const uint4*>(base);
  f.q[1] = *reinterpret_cast<const uint4*>(base + 8);
  return f.v;
}

// B-matrix fragment (32x16 f16, straight K layout), source is row-major B^T
// (rows = N columns of B, contiguous along K): elems 0..15 -> K = kb+16*half+i
__device__ __forceinline__ v16h ldfragB_h(const _Float16* p, int ld, int rb, int kb, int lane) {
  const int r = rb + (lane & 15), hf = lane >> 4;
  const _Float16* base = p + (size_t)r * ld + kb + 16 * hf;
  HFrag f;
  f.q[0] = *reinterpret_cast<const uint4*>(base);
  f.q[1] = *reinterpret_cast<const uint4*>(base + 8);
  return f.v;
}

// B fragment from an f32 row-major B^T source (convert to f16 on the fly)
__device__ __forceinline__ v16h ldfragB_f(const float* p, int ld, int rb, int kb, int lane) {
  const int r = rb + (lane & 15), hf = lane >> 4;
  const float* base = p + (size_t)r * ld + kb + 16 * hf;
  HFrag f;
#pragma unroll
  for (int i = 0; i < 16; ++i) f.e[i] = (_Float16)base[i];
  return f.v;
}

__device__ __forceinline__ v8f wmma16(v16h a, v16h b, v8f c) {
  return __builtin_amdgcn_wmma_f32_16x16x32_f16(false, a, false, b, (short)0, c, false, false);
}

// ---------------------------------------------------------------------------
// Generic tiled GEMM: C[MxN] = A[MxK] @ B[KxN], all f32 row-major, f16 WMMA.
// ---------------------------------------------------------------------------
__global__ void __launch_bounds__(128) gemm_f32_wmma(const float* __restrict__ A,
                                                     const float* __restrict__ B,
                                                     float* __restrict__ C,
                                                     int M, int N, int K) {
  __shared__ _Float16 As[64 * 32];   // A tile, row-major
  __shared__ _Float16 Bts[64 * 32];  // B^T tile: Bts[n][k]
  const int tid = threadIdx.x, lane = tid & 31, wv = tid >> 5;
  const int m0 = blockIdx.x * 64, n0 = blockIdx.y * 64;
  v8f acc[4] = {{}, {}, {}, {}};
  for (int kk = 0; kk < K; kk += 32) {
    for (int i = tid; i < 64 * 32; i += 128) {
      const int r = i >> 5, k = i & 31;
      As[i]  = (_Float16)A[(size_t)(m0 + r) * K + kk + k];
      Bts[i] = (_Float16)B[(size_t)(kk + k) * N + n0 + r];
    }
    if (kk + 32 < K)  // -> global_prefetch_b8
      __builtin_prefetch(A + (size_t)(m0 + (tid >> 1)) * K + kk + 32, 0, 1);
    __syncthreads();
    const v16h a = ldfragA_h(As, 32, wv * 16, 0, lane);
#pragma unroll
    for (int j = 0; j < 4; ++j) {
      const v16h b = ldfragB_h(Bts, 32, j * 16, 0, lane);
      acc[j] = wmma16(a, b, acc[j]);
    }
    __syncthreads();
  }
  const int hf = lane >> 4, nl = lane & 15;
#pragma unroll
  for (int j = 0; j < 4; ++j)
#pragma unroll
    for (int r = 0; r < 8; ++r) {
      const int m = m0 + wv * 16 + r + 8 * hf;
      const int n = n0 + j * 16 + nl;
      C[(size_t)m * N + n] = acc[j][r];
    }
}

// ---------------------------------------------------------------------------
// Taylor feature map: proj[B*L, H*16] f32 -> phi[(b*H+h)*L + l, 288] f16
// Permuted layout: [0..255]=x_i*x_j/(4*sqrt2), [256..271]=x/2, [272]=1, pad 0.
// All stores are 16-byte uint4 writes.
// ---------------------------------------------------------------------------
__global__ void __launch_bounds__(128) taylor_phi_kernel(const float* __restrict__ proj,
                                                         _Float16* __restrict__ phi) {
  const int t = blockIdx.x * 128 + threadIdx.x;   // (b*H+h)*L + l
  const int b = t / (LA_H * LA_L);
  const int h = (t / LA_L) % LA_H;
  const int l = t % LA_L;
  const float* src = proj + (size_t)(b * LA_L + l) * (LA_H * LA_FD) + h * LA_FD;
  _Float16* out = phi + (size_t)t * LA_D;
  float x[LA_FD];
#pragma unroll
  for (int f = 0; f < LA_FD; ++f) x[f] = src[f];
  const float sc = 0.17677669529663687f;  // 1/(sqrt(2)*sqrt(16))
  for (int i = 0; i < LA_FD; ++i) {
    H8 a, bb;
#pragma unroll
    for (int j = 0; j < 8; ++j) { a.e[j] = (_Float16)(x[i] * x[j] * sc); bb.e[j] = (_Float16)(x[i] * x[j + 8] * sc); }
    *reinterpret_cast<uint4*>(out + i * 16)     = a.q;
    *reinterpret_cast<uint4*>(out + i * 16 + 8) = bb.q;
  }
  H8 c0, c1, c2, c3;
#pragma unroll
  for (int j = 0; j < 8; ++j) {
    c0.e[j] = (_Float16)(x[j] * 0.5f);          // rrd = d^{1/4} = 2
    c1.e[j] = (_Float16)(x[j + 8] * 0.5f);
    c2.e[j] = (_Float16)0.0f;
    c3.e[j] = (_Float16)0.0f;
  }
  c2.e[0] = (_Float16)1.0f;                     // the "ones" feature
  *reinterpret_cast<uint4*>(out + 256) = c0.q;
  *reinterpret_cast<uint4*>(out + 264) = c1.q;
  *reinterpret_cast<uint4*>(out + 272) = c2.q;
  *reinterpret_cast<uint4*>(out + 280) = c3.q;
}

// ---------------------------------------------------------------------------
// Batched LDS-tiled transpose: phiK [(bh*L+l)][288] -> phiKT [(bh*288+d)][L]
// ---------------------------------------------------------------------------
__global__ void __launch_bounds__(256) transpose_phik_kernel(const _Float16* __restrict__ in,
                                                             _Float16* __restrict__ out) {
  __shared__ _Float16 t[32][33];
  const int bh = blockIdx.z;
  const int l0 = blockIdx.x * 32, d0 = blockIdx.y * 32;
#pragma unroll
  for (int i = 0; i < 4; ++i) {
    const int l = l0 + threadIdx.y + 8 * i;
    t[threadIdx.y + 8 * i][threadIdx.x] = in[(size_t)(bh * LA_L + l) * LA_D + d0 + threadIdx.x];
  }
  __syncthreads();
#pragma unroll
  for (int i = 0; i < 4; ++i) {
    const int d = d0 + threadIdx.y + 8 * i;
    out[(size_t)(bh * LA_D + d) * LA_L + l0 + threadIdx.x] = t[threadIdx.x][threadIdx.y + 8 * i];
  }
}

// ---------------------------------------------------------------------------
// Batched transpose + f32->f16: vproj [(b*L+l)][768] -> vT [(bh*64+e)][L]
// ---------------------------------------------------------------------------
__global__ void __launch_bounds__(256) transpose_v_kernel(const float* __restrict__ in,
                                                          _Float16* __restrict__ out) {
  __shared__ _Float16 t[32][33];
  const int bh = blockIdx.z;
  const int b = bh / LA_H, h = bh % LA_H;
  const int l0 = blockIdx.x * 32, e0 = blockIdx.y * 32;
#pragma unroll
  for (int i = 0; i < 4; ++i) {
    const int l = l0 + threadIdx.y + 8 * i;
    t[threadIdx.y + 8 * i][threadIdx.x] =
        (_Float16)in[(size_t)(b * LA_L + l) * LA_DM + h * LA_HD + e0 + threadIdx.x];
  }
  __syncthreads();
#pragma unroll
  for (int i = 0; i < 4; ++i) {
    const int e = e0 + threadIdx.y + 8 * i;
    out[(size_t)(bh * LA_HD + e) * LA_L + l0 + threadIdx.x] = t[threadIdx.x][threadIdx.y + 8 * i];
  }
}

// ---------------------------------------------------------------------------
// Chunked causal linear attention. One block per (b,h); 8 waves; sequential
// scan over 16 chunks. State ST[e][d] (f32) lives in LDS; all f16 operands
// are read directly from row-contiguous global layouts with b128 fragments.
// ---------------------------------------------------------------------------
__global__ void __launch_bounds__(256) linattn_kernel(const _Float16* __restrict__ phiQ,
                                                      const _Float16* __restrict__ phiK,
                                                      const _Float16* __restrict__ phiKT,
                                                      const _Float16* __restrict__ vT,
                                                      float* __restrict__ yflat) {
  __shared__ float    ST[LA_HD * LA_D];        // 73728 B : S transposed [e][d]
  __shared__ float    s_vec[LA_D];             //  1152 B : running k-sum
  __shared__ _Float16 innerH[LA_C * LA_C];     // 32768 B : masked q.k^T (f16)
  __shared__ float    den[LA_C];               //   512 B

  const int tid = threadIdx.x, lane = tid & 31, wv = tid >> 5;   // 8 waves
  const int bh = blockIdx.x;                                     // 0..47
  const int b = bh / LA_H, h = bh % LA_H;
  const _Float16* Qbh  = phiQ  + (size_t)bh * LA_L * LA_D;
  const _Float16* Kbh  = phiK  + (size_t)bh * LA_L * LA_D;
  const _Float16* KTbh = phiKT + (size_t)bh * LA_D * LA_L;       // rows = d, contiguous l
  const _Float16* VTbh = vT    + (size_t)bh * LA_HD * LA_L;      // rows = e, contiguous l
  float* Ybh = yflat + (size_t)b * LA_L * LA_DM + h * LA_HD;

  for (int i = tid; i < LA_HD * LA_D; i += 256) ST[i] = 0.0f;
  for (int i = tid; i < LA_D; i += 256) s_vec[i] = 0.0f;
  __syncthreads();

  for (int ch = 0; ch < LA_NCH; ++ch) {
    const _Float16* qC = Qbh + (size_t)ch * LA_C * LA_D;
    const _Float16* kC = Kbh + (size_t)ch * LA_C * LA_D;
    const int cb = ch * LA_C;   // column base into the [.,L] transposed layouts

    // ---- Phase 1: clear innerH; den1[c] = phi_q[c] . s (pre-chunk state) --
    for (int i = tid; i < (LA_C * LA_C) / 8; i += 256)
      reinterpret_cast<uint4*>(innerH)[i] = uint4{0, 0, 0, 0};
    if (tid < LA_C) {
      float d1 = 0.0f;
      const _Float16* qr = qC + (size_t)tid * LA_D;
      for (int d = 0; d < LA_D; d += 8) {
        H8 u; u.q = *reinterpret_cast<const uint4*>(qr + d);
#pragma unroll
        for (int j = 0; j < 8; ++j) d1 += (float)u.e[j] * s_vec[d + j];
      }
      den[tid] = d1;
    }
    __syncthreads();

    // ---- Phase 2: inner = tril(qi . ki^T); den += rowsum ------------------
    for (int t = wv; t < 36; t += 8) {          // lower-triangular 16x16 tiles
      int tm = 0, tt = t;
      while (tt > tm) { tt -= tm + 1; ++tm; }
      const int tn = tt;
      v8f acc = {};
      for (int ks = 0; ks < LA_D / 32; ++ks) {
        const v16h a  = ldfragA_h(qC, LA_D, tm * 16, ks * 32, lane);
        const v16h bb = ldfragB_h(kC, LA_D, tn * 16, ks * 32, lane);
        acc = wmma16(a, bb, acc);
      }
      const int hf = lane >> 4, nl = lane & 15;
#pragma unroll
      for (int r = 0; r < 8; ++r) {
        const int m = tm * 16 + r + 8 * hf;
        const int n = tn * 16 + nl;
        float val = acc[r];
        if (m < n) val = 0.0f;                  // causal mask (diagonal tiles)
        innerH[m * LA_C + n] = (_Float16)val;
        atomicAdd(&den[m], val);
      }
    }
    __syncthreads();

    // ---- Phase 3: num = qi.S + inner.vi ; y = num/(den+eps) ---------------
    for (int t = wv; t < 32; t += 8) {          // 8x4 output tiles (128x64)
      const int tm = t >> 2, te = t & 3;
      v8f acc = {};
      for (int ks = 0; ks < LA_D / 32; ++ks) {
        const v16h a  = ldfragA_h(qC, LA_D, tm * 16, ks * 32, lane);
        const v16h bb = ldfragB_f(ST, LA_D, te * 16, ks * 32, lane);   // S^T rows = e
        acc = wmma16(a, bb, acc);
      }
      for (int cs = 0; cs < LA_C / 32; ++cs) {
        const v16h a  = ldfragA_h(innerH, LA_C, tm * 16, cs * 32, lane);
        const v16h bb = ldfragB_h(VTbh, LA_L, te * 16, cb + cs * 32, lane);  // v^T rows = e
        acc = wmma16(a, bb, acc);
      }
      const int hf = lane >> 4, nl = lane & 15;
#pragma unroll
      for (int r = 0; r < 8; ++r) {
        const int m = tm * 16 + r + 8 * hf;
        const float dd = den[m] + 1e-12f;
        Ybh[(size_t)(cb + m) * LA_DM + te * 16 + nl] = acc[r] / dd;
      }
    }
    __syncthreads();

    // ---- Phase 4: ST[e][d] += sum_c vi[c][e]*ki[c][d] ; s += colsum(ki) ---
    for (int t = wv; t < 72; t += 8) {          // 4x18 tiles of 64x288
      const int tme = t / 18, tnd = t % 18;
      const int hf = lane >> 4, nl = lane & 15;
      v8f acc;
#pragma unroll
      for (int r = 0; r < 8; ++r)
        acc[r] = ST[(size_t)(tme * 16 + r + 8 * hf) * LA_D + tnd * 16 + nl];
      for (int cs = 0; cs < LA_C / 32; ++cs) {
        const v16h a  = ldfragA_h(VTbh, LA_L, tme * 16, cb + cs * 32, lane);  // A[e][c]
        const v16h bb = ldfragB_h(KTbh, LA_L, tnd * 16, cb + cs * 32, lane);  // B^T[d][c]
        acc = wmma16(a, bb, acc);
      }
#pragma unroll
      for (int r = 0; r < 8; ++r)
        ST[(size_t)(tme * 16 + r + 8 * hf) * LA_D + tnd * 16 + nl] = acc[r];
    }
    for (int d = tid; d < LA_D; d += 256) {
      float a = s_vec[d];
      const _Float16* kr = KTbh + (size_t)d * LA_L + cb;
      for (int c = 0; c < LA_C; c += 8) {
        H8 u; u.q = *reinterpret_cast<const uint4*>(kr + c);
#pragma unroll
        for (int j = 0; j < 8; ++j) a += (float)u.e[j];
      }
      s_vec[d] = a;
    }
    __syncthreads();
  }
}

// ---------------------------------------------------------------------------
extern "C" void kernel_launch(void* const* d_in, const int* in_sizes, int n_in,
                              void* d_out, int out_size, void* d_ws, size_t ws_size,
                              hipStream_t stream) {
  (void)in_sizes; (void)n_in; (void)out_size; (void)ws_size;
  const float* hidden = (const float*)d_in[0];
  const float* Wq = (const float*)d_in[1];
  const float* Wk = (const float*)d_in[2];
  const float* Wv = (const float*)d_in[3];
  const float* Wo = (const float*)d_in[4];
  float* out = (float*)d_out;

  char* ws = (char*)d_ws;
  float*    qproj = (float*)(ws + 0);             //  8192 x 192 f32   (6291456 B)
  float*    kproj = (float*)(ws + 6291456);       //  8192 x 192 f32
  float*    vproj = (float*)(ws + 12582912);      //  8192 x 768 f32   (25165824 B)
  _Float16* phiQ  = (_Float16*)(ws + 37748736);   // 98304 x 288 f16   (56623104 B)
  _Float16* phiK  = (_Float16*)(ws + 94371840);   // 98304 x 288 f16
  _Float16* phiKT = (_Float16*)(ws + 150994944);  // 48 x 288 x 2048 f16
  _Float16* vT    = (_Float16*)(ws + 207618048);  // 48 x 64 x 2048 f16 (12582912 B)
  float*    yflat = (float*)(ws + 220200960);     //  8192 x 768 f32

  const int M = LA_B * LA_L;                      // 8192
  dim3 blk(128);
  dim3 tblk(32, 8);
  gemm_f32_wmma<<<dim3(M / 64, (LA_H * LA_FD) / 64), blk, 0, stream>>>(hidden, Wq, qproj, M, LA_H * LA_FD, LA_DM);
  gemm_f32_wmma<<<dim3(M / 64, (LA_H * LA_FD) / 64), blk, 0, stream>>>(hidden, Wk, kproj, M, LA_H * LA_FD, LA_DM);
  gemm_f32_wmma<<<dim3(M / 64, LA_DM / 64), blk, 0, stream>>>(hidden, Wv, vproj, M, LA_DM, LA_DM);
  taylor_phi_kernel<<<dim3((LA_B * LA_H * LA_L) / 128), blk, 0, stream>>>(qproj, phiQ);
  taylor_phi_kernel<<<dim3((LA_B * LA_H * LA_L) / 128), blk, 0, stream>>>(kproj, phiK);
  transpose_phik_kernel<<<dim3(LA_L / 32, LA_D / 32, LA_B * LA_H), tblk, 0, stream>>>(phiK, phiKT);
  transpose_v_kernel<<<dim3(LA_L / 32, LA_HD / 32, LA_B * LA_H), tblk, 0, stream>>>(vproj, vT);
  linattn_kernel<<<dim3(LA_B * LA_H), dim3(256), 0, stream>>>(phiQ, phiK, phiKT, vT, yflat);
  gemm_f32_wmma<<<dim3(M / 64, LA_DM / 64), blk, 0, stream>>>(yflat, Wo, out, M, LA_DM, LA_DM);
}
